// GraphSAGEModel_17944373363173
// MI455X (gfx1250) — matrix operand order repaired and checked by reading, not compiled, not run on hardware
//
#include <hip/hip_runtime.h>

// ---------------------------------------------------------------------------
// GraphSAGE 2-layer forward for MI455X (gfx1250).
//  layer: out = mean_{j->i}(in_j) @ W_l + b + in_i @ W_r   (ReLU after layer 1)
// Strategy:
//  * gather/scatter-mean in bf16 reads + f32 hardware atomics (HBM-bound part)
//  * fused dual-GEMM per layer on v_wmma_f32_16x16x32_bf16, f32 accumulate
//  * weights pre-swizzled once into per-lane-contiguous WMMA B fragments
// ---------------------------------------------------------------------------

typedef __attribute__((ext_vector_type(16))) __bf16 v16bf;
typedef __attribute__((ext_vector_type(4)))  __bf16 v4bf;
typedef __attribute__((ext_vector_type(8)))  float  v8f;

#define N_NODES 100000
#define D_FEAT  128          // d_in == d_hid == 128 (K of both layers)
#define N_STRIPS (N_NODES / 16)   // 6250, exact

union FragAB { v16bf v; uint4 u[2]; };

// ---------------- elementwise helpers ----------------

__global__ void zero_f32(float* __restrict__ p, long n) {
  long i = (long)blockIdx.x * blockDim.x + threadIdx.x;
  if (i < n) p[i] = 0.0f;
}

__global__ void cvt_f32_bf16(const float* __restrict__ in, __bf16* __restrict__ out, long n) {
  long i = (long)blockIdx.x * blockDim.x + threadIdx.x;
  if (i < n) out[i] = (__bf16)in[i];
}

// deg[i] = number of edges with dst == i
__global__ void deg_count(const int* __restrict__ dst, float* __restrict__ deg, int E) {
  int e = blockIdx.x * blockDim.x + threadIdx.x;
  if (e < E) unsafeAtomicAdd(&deg[dst[e]], 1.0f);   // global_atomic_add_f32
}

// agg[dst] += feat[src]; one thread handles 4 features of one edge (8B bf16 load)
__global__ void scatter_add_bf16(const __bf16* __restrict__ feat,
                                 const int* __restrict__ src,
                                 const int* __restrict__ dst,
                                 float* __restrict__ agg, long total) {
  long tid = (long)blockIdx.x * blockDim.x + threadIdx.x;
  if (tid >= total) return;
  int e = (int)(tid >> 5);
  int q = ((int)tid & 31) << 2;          // feature base 0..124
  int s = src[e];
  int d = dst[e];
  v4bf v = *(const v4bf*)(feat + (size_t)s * D_FEAT + q);
  float* ap = agg + (size_t)d * D_FEAT + q;
  unsafeAtomicAdd(ap + 0, (float)v.x);
  unsafeAtomicAdd(ap + 1, (float)v.y);
  unsafeAtomicAdd(ap + 2, (float)v.z);
  unsafeAtomicAdd(ap + 3, (float)v.w);
}

// meanb = bf16(agg / max(deg,1))
__global__ void mean_to_bf16(const float* __restrict__ agg, const float* __restrict__ deg,
                             __bf16* __restrict__ out, long n) {
  long i = (long)blockIdx.x * blockDim.x + threadIdx.x;
  if (i >= n) return;
  int node = (int)(i >> 7);              // /128
  float inv = 1.0f / fmaxf(deg[node], 1.0f);
  out[i] = (__bf16)(agg[i] * inv);
}

// ---------------- weight pre-swizzle into WMMA B fragments ----------------
// B fragment (16x16x32 bf16): lane l supplies N = l&15, K = (l>>4)*16 + e, e=0..15.
// Stored contiguously: frag[((kt*NT + nt)*32 + lane)*16 + e]  -> 32B/lane loads.
__global__ void w_swizzle(const float* __restrict__ W, __bf16* __restrict__ frag,
                          int KT, int NT, int ldw) {
  int t = blockIdx.x * blockDim.x + threadIdx.x;
  int total = KT * NT * 32;
  if (t >= total) return;
  int lane = t & 31;
  int tile = t >> 5;
  int nt = tile % NT;
  int kt = tile / NT;
  int n = nt * 16 + (lane & 15);
  int kbase = kt * 32 + (lane >> 4) * 16;
  __bf16* o = frag + (size_t)t * 16;
#pragma unroll
  for (int e = 0; e < 16; ++e)
    o[e] = (__bf16)W[(size_t)(kbase + e) * ldw + n];
}

// ---------------- fused SAGE GEMM ----------------
// out[row, :] = Am[row,:] @ Wl + bias + Ax[row,:] @ Wr  (optionally ReLU)
// One wave computes a 16-row strip x NT 16-col tiles; K = 128 (4 chunks of 32).
// A fragment (16x32 bf16): lane l, row M = l&15, elements 0..15 -> K = 16*(l>>4)+0..15
//  => two contiguous 16B loads.
// D: element r of v8f -> M = r + 8*(l>>4), N = l&15 within the tile.
template <int NT, bool RELU, typename OutT>
__global__ void sage_gemm(const __bf16* __restrict__ Am, const __bf16* __restrict__ Ax,
                          const __bf16* __restrict__ WlF, const __bf16* __restrict__ WrF,
                          const float* __restrict__ bias, OutT* __restrict__ out,
                          int nstrips) {
  const int lane = threadIdx.x & 31;
  const int wave = threadIdx.x >> 5;
  const int strip = blockIdx.x * (blockDim.x >> 5) + wave;
  if (strip >= nstrips) return;        // wave-uniform: EXEC stays all-ones for WMMA

  const int row = strip * 16 + (lane & 15);
  const __bf16* am = Am + (size_t)row * D_FEAT;
  const __bf16* ax = Ax + (size_t)row * D_FEAT;

  v8f acc[NT];
  v8f zerov = {};
#pragma unroll
  for (int nt = 0; nt < NT; ++nt) acc[nt] = zerov;

  const int KT = D_FEAT / 32;          // 4
#pragma unroll
  for (int kt = 0; kt < KT; ++kt) {
    const int k0 = kt * 32 + ((lane >> 4) << 4);
    FragAB a_m, a_x;
    a_m.u[0] = *(const uint4*)(am + k0);
    a_m.u[1] = *(const uint4*)(am + k0 + 8);
    a_x.u[0] = *(const uint4*)(ax + k0);
    a_x.u[1] = *(const uint4*)(ax + k0 + 8);

    const uint4* wl = (const uint4*)WlF + (size_t)kt * NT * 64 + lane * 2;
    const uint4* wr = (const uint4*)WrF + (size_t)kt * NT * 64 + lane * 2;
#pragma unroll
    for (int nt = 0; nt < NT; ++nt) {
      FragAB bl, br;
      bl.u[0] = wl[nt * 64 + 0];
      bl.u[1] = wl[nt * 64 + 1];
      br.u[0] = wr[nt * 64 + 0];
      br.u[1] = wr[nt * 64 + 1];
      acc[nt] = __builtin_amdgcn_wmma_f32_16x16x32_bf16(
          false, a_m.v, false, bl.v, (short)0, acc[nt], false, false);
      acc[nt] = __builtin_amdgcn_wmma_f32_16x16x32_bf16(
          false, a_x.v, false, br.v, (short)0, acc[nt], false, false);
    }
  }

  const int mbase = strip * 16 + ((lane >> 4) << 3);
  const int ldo = NT * 16;
#pragma unroll
  for (int nt = 0; nt < NT; ++nt) {
    const int col = nt * 16 + (lane & 15);
    const float b = bias[col];
#pragma unroll
    for (int r = 0; r < 8; ++r) {
      float v = acc[nt][r] + b;
      if (RELU) v = fmaxf(v, 0.0f);
      out[(size_t)(mbase + r) * ldo + col] = (OutT)v;
    }
  }
}

// ---------------- host-side orchestration ----------------

static inline size_t align256(size_t x) { return (x + 255) & ~(size_t)255; }

extern "C" void kernel_launch(void* const* d_in, const int* in_sizes, int n_in,
                              void* d_out, int out_size, void* d_ws, size_t ws_size,
                              hipStream_t stream) {
  const float* x    = (const float*)d_in[0];   // [N,128]
  const float* W1l  = (const float*)d_in[1];   // [128,128]
  const float* b1   = (const float*)d_in[2];   // [128]
  const float* W1r  = (const float*)d_in[3];   // [128,128]
  const float* W2l  = (const float*)d_in[4];   // [128,64]
  const float* b2   = (const float*)d_in[5];   // [64]
  const float* W2r  = (const float*)d_in[6];   // [128,64]
  const int*   ei   = (const int*)d_in[7];     // [2,E]
  const int E = in_sizes[7] / 2;
  const int* src = ei;
  const int* dst = ei + E;
  float* out = (float*)d_out;                  // [N,64] f32

  // workspace bump allocation
  char* ws = (char*)d_ws;
  size_t off = 0;
  float*  deg   = (float*)(ws + off);  off = align256(off + (size_t)N_NODES * 4);
  float*  agg   = (float*)(ws + off);  off = align256(off + (size_t)N_NODES * D_FEAT * 4);
  __bf16* xb    = (__bf16*)(ws + off); off = align256(off + (size_t)N_NODES * D_FEAT * 2);
  __bf16* hb    = (__bf16*)(ws + off); off = align256(off + (size_t)N_NODES * D_FEAT * 2);
  __bf16* meanb = (__bf16*)(ws + off); off = align256(off + (size_t)N_NODES * D_FEAT * 2);
  __bf16* w1lF  = (__bf16*)(ws + off); off = align256(off + 128 * 128 * 2);
  __bf16* w1rF  = (__bf16*)(ws + off); off = align256(off + 128 * 128 * 2);
  __bf16* w2lF  = (__bf16*)(ws + off); off = align256(off + 128 * 64 * 2);
  __bf16* w2rF  = (__bf16*)(ws + off); off = align256(off + 128 * 64 * 2);

  const long nfeat = (long)N_NODES * D_FEAT;    // 12.8M
  const int  TPB = 256;
  dim3 blk(TPB);

  // init: deg=0, agg=0
  zero_f32<<<dim3((N_NODES + TPB - 1) / TPB), blk, 0, stream>>>(deg, N_NODES);
  zero_f32<<<dim3((unsigned)((nfeat + TPB - 1) / TPB)), blk, 0, stream>>>(agg, nfeat);

  // bf16 copy of x; swizzled bf16 weights
  cvt_f32_bf16<<<dim3((unsigned)((nfeat + TPB - 1) / TPB)), blk, 0, stream>>>(x, xb, nfeat);
  w_swizzle<<<dim3((4 * 8 * 32 + TPB - 1) / TPB), blk, 0, stream>>>(W1l, w1lF, 4, 8, 128);
  w_swizzle<<<dim3((4 * 8 * 32 + TPB - 1) / TPB), blk, 0, stream>>>(W1r, w1rF, 4, 8, 128);
  w_swizzle<<<dim3((4 * 4 * 32 + TPB - 1) / TPB), blk, 0, stream>>>(W2l, w2lF, 4, 4, 64);
  w_swizzle<<<dim3((4 * 4 * 32 + TPB - 1) / TPB), blk, 0, stream>>>(W2r, w2rF, 4, 4, 64);

  // degrees (shared by both layers)
  deg_count<<<dim3((E + TPB - 1) / TPB), blk, 0, stream>>>(dst, deg, E);

  // ---- layer 1 ----
  const long sc_total = (long)E * 32;
  scatter_add_bf16<<<dim3((unsigned)((sc_total + TPB - 1) / TPB)), blk, 0, stream>>>(
      xb, src, dst, agg, sc_total);
  mean_to_bf16<<<dim3((unsigned)((nfeat + TPB - 1) / TPB)), blk, 0, stream>>>(
      agg, deg, meanb, nfeat);
  {
    const int waves = TPB / 32;
    dim3 grid((N_STRIPS + waves - 1) / waves);
    sage_gemm<8, true, __bf16><<<grid, blk, 0, stream>>>(
        meanb, xb, w1lF, w1rF, b1, hb, N_STRIPS);
  }

  // ---- layer 2 ----
  zero_f32<<<dim3((unsigned)((nfeat + TPB - 1) / TPB)), blk, 0, stream>>>(agg, nfeat);
  scatter_add_bf16<<<dim3((unsigned)((sc_total + TPB - 1) / TPB)), blk, 0, stream>>>(
      hb, src, dst, agg, sc_total);
  mean_to_bf16<<<dim3((unsigned)((nfeat + TPB - 1) / TPB)), blk, 0, stream>>>(
      agg, deg, meanb, nfeat);
  {
    const int waves = TPB / 32;
    dim3 grid((N_STRIPS + waves - 1) / waves);
    sage_gemm<4, false, float><<<grid, blk, 0, stream>>>(
        meanb, hb, w2lF, w2rF, b2, out, N_STRIPS);
  }
}
